// WindowedAttention_37984690766439
// MI455X (gfx1250) — compile-verified, matrix operand order checked by reading, not run on hardware
//
#include <hip/hip_runtime.h>
#include <cstdint>
#include <cstddef>

typedef __attribute__((ext_vector_type(16))) _Float16 v16h;
typedef __attribute__((ext_vector_type(8)))  _Float16 v8h;
typedef __attribute__((ext_vector_type(8)))  float    v8f;
typedef __attribute__((ext_vector_type(4)))  int      v4i;

#define S_TOK 2304
#define DIMW  1024
#define NHEAD 16
#define HDIM  64
#define TBL_N 9025   // (2*48-1)^2

#if __has_builtin(__builtin_amdgcn_global_load_async_to_lds_b128)
#define HAVE_ASYNC_LDS 1
typedef __attribute__((address_space(1))) v4i gv4i;   // global (device) 16B
typedef __attribute__((address_space(3))) v4i lv4i;   // LDS 16B
#else
#define HAVE_ASYNC_LDS 0
#endif

// Stage 16 bytes/lane from global into LDS. CDNA5 path: async DMA (ASYNCcnt),
// no VGPR round-trip. Fallback: load to regs + ds_store.
__device__ __forceinline__ void stage16B(const _Float16* g, _Float16* l) {
#if HAVE_ASYNC_LDS
  __builtin_amdgcn_global_load_async_to_lds_b128((gv4i*)g, (lv4i*)l, 0, 0);
#else
  *(v8h*)l = *(const v8h*)g;
#endif
}

__device__ __forceinline__ void async_stage_wait() {
#if HAVE_ASYNC_LDS
#if __has_builtin(__builtin_amdgcn_s_wait_asynccnt)
  __builtin_amdgcn_s_wait_asynccnt(0);
#else
  asm volatile("s_wait_asynccnt 0x0" ::: "memory");
#endif
#endif
}

__device__ __forceinline__ v8f wmma_f16(v16h a, v16h b, v8f c) {
  // D = A(16x32 f16) * B(32x16 f16) + C(16x16 f32)
  return __builtin_amdgcn_wmma_f32_16x16x32_f16(
      /*neg_a=*/false, a, /*neg_b=*/false, b,
      /*c_mod=*/(short)0, c, /*reuse_a=*/false, /*reuse_b=*/false);
}

// ---------------------------------------------------------------- casts
__global__ void swin_cast_f16(const float* __restrict__ in,
                              _Float16* __restrict__ out, int n) {
  int i = blockIdx.x * 256 + threadIdx.x;
  if (i < n) out[i] = (_Float16)in[i];
}

// ---------------------------------------------------------------- CPB MLP
// bias[r][h] = relu(table[r] @ w1^T + b1) @ w2[h];  sig16 = 16*sigmoid(bias)
__global__ __launch_bounds__(256) void swin_cpb(
    const float* __restrict__ table, const float* __restrict__ w1,
    const float* __restrict__ b1,    const float* __restrict__ w2,
    float* __restrict__ sig16) {
  __shared__ float red[256 * NHEAD];
  const int r = blockIdx.x;
  const int t = threadIdx.x;
  const float t0 = table[r * 2 + 0];
  const float t1 = table[r * 2 + 1];
  float acc[NHEAD];
#pragma unroll
  for (int h = 0; h < NHEAD; ++h) acc[h] = 0.f;
  for (int j = t; j < 512; j += 256) {
    float hv = fmaxf(0.f, t0 * w1[j * 2 + 0] + t1 * w1[j * 2 + 1] + b1[j]);
#pragma unroll
    for (int h = 0; h < NHEAD; ++h) acc[h] += hv * w2[h * 512 + j];
  }
#pragma unroll
  for (int h = 0; h < NHEAD; ++h) red[t * NHEAD + h] = acc[h];
  __syncthreads();
  for (int st = 128; st >= 1; st >>= 1) {
    if (t < st) {
#pragma unroll
      for (int h = 0; h < NHEAD; ++h)
        red[t * NHEAD + h] += red[(t + st) * NHEAD + h];
    }
    __syncthreads();
  }
  if (t < NHEAD) {
    float b = red[t];
    sig16[r * NHEAD + t] = 16.f / (1.f + expf(-b));
  }
}

// ---------------------------------------------------------------- QKV GEMM
// C[m,n] = sum_k xh[m,k]*wh[n,k] + qkv_b[n]; scatter into qh/kh/vT (f16).
// Block tile 128(M)x64(N); B tile (64x32 f16) async-staged into LDS, double
// buffered, shared by all 8 waves. A fragment register-prefetched 1 step ahead.
__global__ __launch_bounds__(256) void swin_gemm_qkv(
    const _Float16* __restrict__ xh, const _Float16* __restrict__ wh,
    const float* __restrict__ qkv_b,
    _Float16* __restrict__ qh, _Float16* __restrict__ kh,
    _Float16* __restrict__ vT) {
  __shared__ _Float16 Bs[2][64 * 32];
  const int lane = threadIdx.x & 31;
  const int wv   = threadIdx.x >> 5;
  const int ln16 = lane & 15;
  const int hi   = lane >> 4;        // 0 or 1
  const int kh16 = hi << 4;          // K-half offset: 0 or 16
  const int m0 = blockIdx.y * 128 + wv * 16;
  const int n0 = blockIdx.x * 64;

  const int brow = threadIdx.x >> 2;          // 0..63
  const int bcg  = (threadIdx.x & 3) * 8;     // 0,8,16,24
  const _Float16* bSrc = wh + (size_t)(n0 + brow) * DIMW + bcg;
  _Float16* bDst = &Bs[0][brow * 32 + bcg];   // buffer 1 is +64*32 halves

  const _Float16* aRow = xh + (size_t)(m0 + ln16) * DIMW + kh16;
  stage16B(bSrc, bDst);
  v16h a_cur = *(const v16h*)(aRow);
  v8f acc[4] = {};

  for (int ks = 0; ks < DIMW / 32; ++ks) {
    async_stage_wait();                    // our async writes to LDS landed
    __syncthreads();                       // stage ks visible to all waves
    const int cur = ks & 1;
    v16h a_nxt = a_cur;
    if (ks + 1 < DIMW / 32) {              // overlap with this step's WMMAs
      stage16B(bSrc + (ks + 1) * 32, bDst + (cur ^ 1) * (64 * 32));
      a_nxt = *(const v16h*)(aRow + (ks + 1) * 32);
    }
    v16h bfr[4];
#pragma unroll
    for (int t = 0; t < 4; ++t)
      bfr[t] = *(const v16h*)&Bs[cur][(16 * t + ln16) * 32 + kh16];
#pragma unroll
    for (int t = 0; t < 4; ++t)
      acc[t] = wmma_f16(a_cur, bfr[t], acc[t]);
    a_cur = a_nxt;
  }

  // Epilogue. The 64-column tile lies entirely inside Q, K, or V (region
  // boundary 1024 is a multiple of 64) -> block-uniform routing.
  const int mBase = m0 + hi * 8;
  if (n0 >= 2 * DIMW) {
    // V region, transposed store vT[h][d][s]: lane holds fixed d and 8
    // consecutive tokens -> one 16B packed store per tile.
#pragma unroll
    for (int t = 0; t < 4; ++t) {
      const int n = n0 + 16 * t + ln16;
      const float bn = qkv_b[n];
      const int nn = n - 2 * DIMW, h = nn >> 6, d = nn & 63;
      v8h pk;
#pragma unroll
      for (int r = 0; r < 8; ++r) pk[r] = (_Float16)(acc[t][r] + bn);
      *(v8h*)(vT + ((size_t)h * HDIM + d) * S_TOK + mBase) = pk;
    }
  } else {
#pragma unroll
    for (int t = 0; t < 4; ++t) {
      const int n = n0 + 16 * t + ln16;
      const float bn = qkv_b[n];
#pragma unroll
      for (int r = 0; r < 8; ++r) {
        const int m = mBase + r;
        float c = acc[t][r] + bn;
        if (n < DIMW) {                            // Q (pre-scale by 1/sqrt(64))
          int h = n >> 6, d = n & 63;
          qh[((size_t)h * S_TOK + m) * HDIM + d] = (_Float16)(c * 0.125f);
        } else {                                   // K
          int nn = n - DIMW, h = nn >> 6, d = nn & 63;
          kh[((size_t)h * S_TOK + m) * HDIM + d] = (_Float16)c;
        }
      }
    }
  }
}

// ---------------------------------------------------------------- fused attention
// scoresT = K·Qᵀ (M=key, N=query) so each lane owns a fixed query column;
// online softmax; outᵀ = Vᵀ·P via WMMA; P fragment rebuilt with shfl_xor(16).
// K-tile (32x64) and Vᵀ-tile (64x32) async-staged into LDS (double buffered),
// shared by all 8 waves of the block. grid (2304/128, 16 heads).
__global__ __launch_bounds__(256) void swin_attn(
    const _Float16* __restrict__ qh, const _Float16* __restrict__ kh,
    const _Float16* __restrict__ vT, const float* __restrict__ sig16,
    const int* __restrict__ rpi, _Float16* __restrict__ ah) {
  __shared__ _Float16 Ks[2][32 * 64];
  __shared__ _Float16 Vs[2][64 * 32];
  const int lane = threadIdx.x & 31;
  const int wv   = threadIdx.x >> 5;
  const int ln16 = lane & 15;
  const int hi   = lane >> 4;
  const int kh16 = hi << 4;
  const int h  = blockIdx.y;
  const int q0 = blockIdx.x * 128 + wv * 16;
  const int q  = q0 + ln16;                  // this lane's query column

  const _Float16* Q = qh + (size_t)h * S_TOK * HDIM;
  const _Float16* K = kh + (size_t)h * S_TOK * HDIM;
  const _Float16* V = vT + (size_t)h * HDIM * S_TOK;
  const int* rowIdx = rpi + (size_t)q * S_TOK;   // loop-invariant per lane

  const int krow = threadIdx.x >> 3;          // 0..31 (key row)
  const int kcg  = (threadIdx.x & 7) * 8;     // 0..56 (d)
  const int vrow = threadIdx.x >> 2;          // 0..63 (d row)
  const int vcg  = (threadIdx.x & 3) * 8;     // 0..24 (key)
  const _Float16* kSrc = K + (size_t)krow * HDIM + kcg;
  const _Float16* vSrc = V + (size_t)vrow * S_TOK + vcg;
  _Float16* kDst = &Ks[0][krow * 64 + kcg];
  _Float16* vDst = &Vs[0][vrow * 32 + vcg];

  // Q fragments as WMMA B operands (lane = query column, 16 contiguous d)
  const v16h bq0 = *(const v16h*)(Q + (size_t)q * HDIM + kh16);        // d 0..31
  const v16h bq1 = *(const v16h*)(Q + (size_t)q * HDIM + 32 + kh16);   // d 32..63

  stage16B(kSrc, kDst);
  stage16B(vSrc, vDst);
  v8f o[4] = {};
  float m_run = -INFINITY, l_run = 0.f;

  for (int kb = 0; kb < S_TOK; kb += 32) {
    async_stage_wait();
    __syncthreads();
    const int cur = (kb >> 5) & 1;
    if (kb + 32 < S_TOK) {
      stage16B(kSrc + (size_t)(kb + 32) * HDIM, kDst + (cur ^ 1) * (32 * 64));
      stage16B(vSrc + (kb + 32),                vDst + (cur ^ 1) * (64 * 32));
    }
    if (kb + 64 < S_TOK)                      // pull next idx row segment to L2/L0
      __builtin_prefetch(rowIdx + kb + 64, 0, 1);

    // --- scoresT: two 16(key)x16(query) tiles, K-dim = 64 (2 WMMAs each)
    v8f c0 = {}, c1 = {};
    {
      v16h a00 = *(const v16h*)&Ks[cur][(ln16)      * 64 + kh16];
      v16h a01 = *(const v16h*)&Ks[cur][(ln16)      * 64 + 32 + kh16];
      v16h a10 = *(const v16h*)&Ks[cur][(16 + ln16) * 64 + kh16];
      v16h a11 = *(const v16h*)&Ks[cur][(16 + ln16) * 64 + 32 + kh16];
      c0 = wmma_f16(a00, bq0, c0);
      c0 = wmma_f16(a01, bq1, c0);
      c1 = wmma_f16(a10, bq0, c1);
      c1 = wmma_f16(a11, bq1, c1);
    }

    // --- bias gather (L2-resident table) + block max
    float s0[8], s1[8];
    const int kb0 = kb + hi * 8;        // keys this lane owns in tile0
    const int kb1 = kb + 16 + hi * 8;   // keys this lane owns in tile1
    float mx = -INFINITY;
#pragma unroll
    for (int r = 0; r < 8; ++r) {
      s0[r] = c0[r] + sig16[rowIdx[kb0 + r] * NHEAD + h];
      s1[r] = c1[r] + sig16[rowIdx[kb1 + r] * NHEAD + h];
      mx = fmaxf(mx, fmaxf(s0[r], s1[r]));
    }
    mx = fmaxf(mx, __shfl_xor(mx, 16, 32));   // combine lane pair (same query)

    // --- online softmax update
    const float m_new = fmaxf(m_run, mx);
    const float scale = expf(m_run - m_new);
    float ls = 0.f;
#pragma unroll
    for (int r = 0; r < 8; ++r) {
      s0[r] = expf(s0[r] - m_new);
      s1[r] = expf(s1[r] - m_new);
      ls += s0[r] + s1[r];
    }
    ls += __shfl_xor(ls, 16, 32);
    l_run = l_run * scale + ls;
    m_run = m_new;
#pragma unroll
    for (int t = 0; t < 4; ++t) o[t] = o[t] * scale;   // per-lane (per-query) rescale

    // --- rebuild P as a WMMA B fragment (32 keys x 16 queries) via shuffles:
    // lane<16 needs keys kb..kb+15 (tile0), lane>=16 needs keys kb+16..kb+31 (tile1)
    v16h pf;
#pragma unroll
    for (int e = 0; e < 8; ++e) {
      const float p0sw = __shfl_xor(s0[e], 16, 32);
      const float p1sw = __shfl_xor(s1[e], 16, 32);
      const float lo = hi ? p1sw : s0[e];   // K_rel 0..7
      const float up = hi ? s1[e] : p0sw;   // K_rel 8..15
      pf[e]     = (_Float16)lo;
      pf[e + 8] = (_Float16)up;
    }

    // --- outᵀ += Vᵀ(16d x 32k) · P(32k x 16q), 4 d-tiles
    v16h av[4];
#pragma unroll
    for (int t = 0; t < 4; ++t)
      av[t] = *(const v16h*)&Vs[cur][(16 * t + ln16) * 32 + kh16];
#pragma unroll
    for (int t = 0; t < 4; ++t)
      o[t] = wmma_f16(av[t], pf, o[t]);
  }

  // Lane holds fixed q and 8 consecutive d -> one 16B packed store per d-tile.
  const float inv = 1.f / l_run;
  _Float16* outBase = ah + (size_t)q * DIMW + h * HDIM + hi * 8;
#pragma unroll
  for (int t = 0; t < 4; ++t) {
    v8h pk;
#pragma unroll
    for (int r = 0; r < 8; ++r) pk[r] = (_Float16)(o[t][r] * inv);
    *(v8h*)(outBase + 16 * t) = pk;
  }
}

// ---------------------------------------------------------------- output projection
// out[m,n] = sum_k ah[m,k]*pwh[n,k] + proj_b[n]   (f32 result)
__global__ __launch_bounds__(256) void swin_gemm_proj(
    const _Float16* __restrict__ ah, const _Float16* __restrict__ pwh,
    const float* __restrict__ proj_b, float* __restrict__ out) {
  __shared__ _Float16 Bs[2][64 * 32];
  const int lane = threadIdx.x & 31;
  const int wv   = threadIdx.x >> 5;
  const int ln16 = lane & 15;
  const int hi   = lane >> 4;
  const int kh16 = hi << 4;
  const int m0 = blockIdx.y * 128 + wv * 16;
  const int n0 = blockIdx.x * 64;

  const int brow = threadIdx.x >> 2;
  const int bcg  = (threadIdx.x & 3) * 8;
  const _Float16* bSrc = pwh + (size_t)(n0 + brow) * DIMW + bcg;
  _Float16* bDst = &Bs[0][brow * 32 + bcg];

  const _Float16* aRow = ah + (size_t)(m0 + ln16) * DIMW + kh16;
  stage16B(bSrc, bDst);
  v16h a_cur = *(const v16h*)(aRow);
  v8f acc[4] = {};

  for (int ks = 0; ks < DIMW / 32; ++ks) {
    async_stage_wait();
    __syncthreads();
    const int cur = ks & 1;
    v16h a_nxt = a_cur;
    if (ks + 1 < DIMW / 32) {
      stage16B(bSrc + (ks + 1) * 32, bDst + (cur ^ 1) * (64 * 32));
      a_nxt = *(const v16h*)(aRow + (ks + 1) * 32);
    }
    v16h bfr[4];
#pragma unroll
    for (int t = 0; t < 4; ++t)
      bfr[t] = *(const v16h*)&Bs[cur][(16 * t + ln16) * 32 + kh16];
#pragma unroll
    for (int t = 0; t < 4; ++t)
      acc[t] = wmma_f16(a_cur, bfr[t], acc[t]);
    a_cur = a_nxt;
  }

  const int mBase = m0 + hi * 8;
#pragma unroll
  for (int t = 0; t < 4; ++t) {
    const int n = n0 + 16 * t + ln16;
    const float bn = proj_b[n];
#pragma unroll
    for (int r = 0; r < 8; ++r)
      out[(size_t)(mBase + r) * DIMW + n] = acc[t][r] + bn;
  }
}

// ---------------------------------------------------------------- launch
extern "C" void kernel_launch(void* const* d_in, const int* in_sizes, int n_in,
                              void* d_out, int out_size, void* d_ws, size_t ws_size,
                              hipStream_t stream) {
  (void)in_sizes; (void)n_in; (void)out_size; (void)ws_size;
  const float* x      = (const float*)d_in[0];
  const float* qkv_w  = (const float*)d_in[1];
  const float* qkv_b  = (const float*)d_in[2];
  const float* proj_w = (const float*)d_in[3];
  const float* proj_b = (const float*)d_in[4];
  const float* cpb_w1 = (const float*)d_in[5];
  const float* cpb_b1 = (const float*)d_in[6];
  const float* cpb_w2 = (const float*)d_in[7];
  const float* tbl    = (const float*)d_in[8];
  const int*   rpi    = (const int*)d_in[9];
  float* out = (float*)d_out;

  uintptr_t p = (uintptr_t)d_ws;
  auto carve = [&](size_t bytes) -> void* {
    p = (p + 255) & ~(uintptr_t)255;
    void* r = (void*)p;
    p += bytes;
    return r;
  };
  _Float16* xh    = (_Float16*)carve((size_t)S_TOK * DIMW * 2);
  _Float16* wh    = (_Float16*)carve((size_t)3 * DIMW * DIMW * 2);
  _Float16* pwh   = (_Float16*)carve((size_t)DIMW * DIMW * 2);
  _Float16* qh    = (_Float16*)carve((size_t)S_TOK * DIMW * 2);
  _Float16* kh    = (_Float16*)carve((size_t)S_TOK * DIMW * 2);
  _Float16* vT    = (_Float16*)carve((size_t)S_TOK * DIMW * 2);
  _Float16* ahbuf = (_Float16*)carve((size_t)S_TOK * DIMW * 2);
  float*    sig16 = (float*)carve((size_t)TBL_N * NHEAD * 4);

  const int nx = S_TOK * DIMW;
  const int nw = 3 * DIMW * DIMW;
  const int np = DIMW * DIMW;
  swin_cast_f16<<<(nx + 255) / 256, 256, 0, stream>>>(x, xh, nx);
  swin_cast_f16<<<(nw + 255) / 256, 256, 0, stream>>>(qkv_w, wh, nw);
  swin_cast_f16<<<(np + 255) / 256, 256, 0, stream>>>(proj_w, pwh, np);

  swin_cpb<<<TBL_N, 256, 0, stream>>>(tbl, cpb_w1, cpb_b1, cpb_w2, sig16);

  swin_gemm_qkv<<<dim3(3 * DIMW / 64, S_TOK / 128), 256, 0, stream>>>(
      xh, wh, qkv_b, qh, kh, vT);

  swin_attn<<<dim3(S_TOK / 128, NHEAD), 256, 0, stream>>>(
      qh, kh, vT, sig16, rpi, ahbuf);

  swin_gemm_proj<<<dim3(DIMW / 64, S_TOK / 128), 256, 0, stream>>>(
      ahbuf, pwh, proj_b, out);
}
